// LSTMWithAttention_55825984913848
// MI455X (gfx1250) — matrix operand order repaired and checked by reading, not compiled, not run on hardware
//
#include <hip/hip_runtime.h>
#include <hip/hip_bf16.h>

// ---------------------------------------------------------------------------
// LSTM(7 layers) + attention head for MI455X (gfx1250, wave32, WMMA).
//
//  * x_to_bf16_kernel: one streaming fp32->bf16 pass so the serialized scan
//    never converts (layer0 previously burned 16 v_cvt_pk_bf16_f32 per step
//    inside the recurrence).
//  * lstm_scan: persistent fused kernel per layer. grid=8 (B/16), block=256
//    (8 waves). Wave w owns H-columns [16w,16w+16) of all 4 gates. Wih/Whh
//    live as bf16 WMMA B-fragments in registers (~256 VGPRs) for the whole
//    T loop; cell state c in C-fragment registers; h roundtrips through a
//    bank-padded LDS tile. z = bias + x_t@Wih^T + h@Whh^T via 32x
//    v_wmma_f32_16x16x32_bf16 per wave per step. global_prefetch for t+2.
//  * attn_final_kernel: only attention row A-1 reaches the output and
//    softmax is per-(b,a) over T, so attention collapses to one
//    softmax-weighted time reduction. Plain VALU (67 MFLOP).
// ---------------------------------------------------------------------------

typedef __attribute__((ext_vector_type(16))) __bf16 v16bf;
typedef __attribute__((ext_vector_type(8)))  __bf16 v8bf;
typedef __attribute__((ext_vector_type(8)))  float  v8f;

#define T_DIM 1024
#define H_DIM 128
#define B_DIM 128
#define A_DIM 128
#define L_DIM 7

__device__ __forceinline__ unsigned short f2bf(float x) {
  return __builtin_bit_cast(unsigned short, (__bf16)x);
}
__device__ __forceinline__ float bf2f(unsigned short u) {
  return (float)__builtin_bit_cast(__bf16, u);
}
__device__ __forceinline__ float fsig(float x) {
  return 1.0f / (1.0f + __expf(-x));
}
__device__ __forceinline__ float ftanh(float x) {
  x = fminf(12.0f, fmaxf(-12.0f, x));
  float e = __expf(2.0f * x);
  return (e - 1.0f) / (e + 1.0f);
}

// A-fragment (16-bit 16x32, ISA 7.12.2): lane(0-15)=row M, element j:
//   j<8  -> K = kt*32 + (lane>=16?8:0)  + j
//   j>=8 -> K = kt*32 + 16 + (lane>=16?8:0) + (j-8)
// Two contiguous 16B runs per lane -> b128 loads.
__device__ __forceinline__ v16bf load_a_bf(const unsigned short* rowp, int kt, int lhi) {
  const int base = kt * 32 + lhi * 8;
  v8bf lo = *(const v8bf*)(const void*)(rowp + base);
  v8bf hi = *(const v8bf*)(const void*)(rowp + base + 16);
  v16bf a;
#pragma unroll
  for (int j = 0; j < 8; ++j) { a[j] = lo[j]; a[8 + j] = hi[j]; }
  return a;
}

// B-fragment (16-bit 32x16): lane(0-15)=col N, element j -> K = kt*32 +
// (lane>=16?8:0) + (j<8 ? j : 16+(j-8)). B[k][n] = W[n][k] (same order as A).
__device__ __forceinline__ v16bf make_b_frag(const float* W, int nrow, int kt, int lhi) {
  const float* wr = W + (size_t)nrow * H_DIM + kt * 32 + lhi * 8;
  v16bf b;
#pragma unroll
  for (int j = 0; j < 16; ++j) b[j] = (__bf16)wr[j < 8 ? j : j + 8];
  return b;
}

__global__ __launch_bounds__(256) void x_to_bf16_kernel(
    const float* __restrict__ in, unsigned short* __restrict__ out, int n) {
  int i = (blockIdx.x * blockDim.x + threadIdx.x) * 4;
  if (i + 3 < n) {
    const float4 v = *(const float4*)(in + i);
    unsigned short o0 = f2bf(v.x), o1 = f2bf(v.y), o2 = f2bf(v.z), o3 = f2bf(v.w);
    ushort4 o; o.x = o0; o.y = o1; o.z = o2; o.w = o3;
    *(ushort4*)(out + i) = o;
  }
}

__global__ __launch_bounds__(256, 1) void lstm_scan(
    const unsigned short* __restrict__ in_seq,  // [B,T,H] bf16
    unsigned short* __restrict__ out_seq,       // [B,T,H] bf16
    const float* __restrict__ Wih,              // [512,128]
    const float* __restrict__ Whh,              // [512,128]
    const float* __restrict__ bih,              // [512]
    const float* __restrict__ bhh)              // [512]
{
  const int wv    = threadIdx.x >> 5;   // wave 0..7 -> H column tile
  const int lane  = threadIdx.x & 31;
  const int l15   = lane & 15;
  const int lhi   = lane >> 4;
  const int bbase = blockIdx.x * 16;    // batch tile
  const int rbase = lhi ? 8 : 0;        // C-fragment row base

  __shared__ __align__(16) unsigned short hbuf[16][H_DIM + 8];  // +8 bf16 pad

  // Resident weight fragments: gate g (i,f,g,o), k-tile kt.
  v16bf BIH[4][4], BHH[4][4];
  float bias_s[4];
#pragma unroll
  for (int g = 0; g < 4; ++g) {
    const int nrow = g * H_DIM + wv * 16 + l15;  // output neuron (col N)
    bias_s[g] = bih[nrow] + bhh[nrow];
#pragma unroll
    for (int kt = 0; kt < 4; ++kt) {
      BIH[g][kt] = make_b_frag(Wih, nrow, kt, lhi);
      BHH[g][kt] = make_b_frag(Whh, nrow, kt, lhi);
    }
  }

  v8f cfrag = {};  // cell state, C-fragment layout
  const unsigned short* xrow = in_seq + (size_t)(bbase + l15) * T_DIM * H_DIM;
  const int col = wv * 16 + l15;

  for (int t = 0; t < T_DIM; ++t) {
    // Pull t+2's row toward the WGP cache (global_prefetch_b8); the row is
    // 256B => two cachelines per lane-pair.
    if (t + 2 < T_DIM) {
      const unsigned short* np = xrow + (size_t)(t + 2) * H_DIM;
      __builtin_prefetch((const void*)np, 0, 1);
      __builtin_prefetch((const void*)(np + 64), 0, 1);
    }

    // ---- input-projection part (independent of h; hides barrier latency)
    v16bf xA[4];
    const unsigned short* xp = xrow + (size_t)t * H_DIM;
#pragma unroll
    for (int kt = 0; kt < 4; ++kt) xA[kt] = load_a_bf(xp, kt, lhi);

    v8f z[4];
#pragma unroll
    for (int g = 0; g < 4; ++g) {
      v8f acc;
#pragma unroll
      for (int r = 0; r < 8; ++r) acc[r] = bias_s[g];
#pragma unroll
      for (int kt = 0; kt < 4; ++kt)
        acc = __builtin_amdgcn_wmma_f32_16x16x32_bf16(false, xA[kt], false,
                                                      BIH[g][kt], (short)0, acc,
                                                      false, false);
      z[g] = acc;
    }

    // ---- recurrent part (h_{t-1} from LDS; h_0 == 0 so skipped at t==0)
    if (t > 0) {
      v16bf hA[4];
      const unsigned short* hp = &hbuf[l15][0];
#pragma unroll
      for (int kt = 0; kt < 4; ++kt) hA[kt] = load_a_bf(hp, kt, lhi);
#pragma unroll
      for (int g = 0; g < 4; ++g) {
        v8f acc = z[g];
#pragma unroll
        for (int kt = 0; kt < 4; ++kt)
          acc = __builtin_amdgcn_wmma_f32_16x16x32_bf16(false, hA[kt], false,
                                                        BHH[g][kt], (short)0, acc,
                                                        false, false);
        z[g] = acc;
      }
    }

    // ---- elementwise gates in C-fragment layout (i,f,g,o = z[0..3])
    v8f hf;
#pragma unroll
    for (int r = 0; r < 8; ++r) {
      float iv = fsig(z[0][r]);
      float fv = fsig(z[1][r]);
      float gv = ftanh(z[2][r]);
      float ov = fsig(z[3][r]);
      float cv = fv * cfrag[r] + iv * gv;
      cfrag[r] = cv;
      hf[r] = ov * ftanh(cv);
    }

    __syncthreads();  // all waves finished reading h_{t-1}
#pragma unroll
    for (int r = 0; r < 8; ++r) {
      unsigned short hb = f2bf(hf[r]);
      hbuf[rbase + r][col] = hb;
      out_seq[(size_t)(bbase + rbase + r) * T_DIM * H_DIM + (size_t)t * H_DIM + col] = hb;
    }
    __syncthreads();  // h_t visible to all waves
  }
}

// Only attention row A-1 reaches the output; softmax over T is per-(b,a),
// so the head is: p = softmax_t(tanh(out . Wa[A-1] + ba[A-1])),
// logit = (sum_t p_t * out[b,t,:]) . Wf + bf.
__global__ __launch_bounds__(128) void attn_final_kernel(
    const unsigned short* __restrict__ hseq,  // [B,T,H] bf16
    const float* __restrict__ Wa, const float* __restrict__ ba,
    const float* __restrict__ Wf, const float* __restrict__ bfp,
    float* __restrict__ outp)
{
  const int b = blockIdx.x, tid = threadIdx.x;
  __shared__ float wa[H_DIM], wf[H_DIM], sbuf[T_DIM], red[128];
  wa[tid] = Wa[(A_DIM - 1) * H_DIM + tid];
  wf[tid] = Wf[tid];
  __syncthreads();

  const unsigned short* base = hseq + (size_t)b * T_DIM * H_DIM;
  const float ba_last = ba[A_DIM - 1];

  float lmax = -3.0e38f;
  for (int tt = tid; tt < T_DIM; tt += 128) {
    const unsigned short* row = base + (size_t)tt * H_DIM;
    float acc = 0.f;
#pragma unroll
    for (int h = 0; h < H_DIM; h += 8) {
      v8bf v = *(const v8bf*)(const void*)(row + h);
#pragma unroll
      for (int j = 0; j < 8; ++j) acc += (float)v[j] * wa[h + j];
    }
    float sv = ftanh(acc + ba_last);
    sbuf[tt] = sv;
    lmax = fmaxf(lmax, sv);
  }
  red[tid] = lmax; __syncthreads();
  for (int st = 64; st > 0; st >>= 1) {
    if (tid < st) red[tid] = fmaxf(red[tid], red[tid + st]);
    __syncthreads();
  }
  float mx = red[0];
  __syncthreads();

  float lsum = 0.f;
  for (int tt = tid; tt < T_DIM; tt += 128) {
    float p = __expf(sbuf[tt] - mx);
    sbuf[tt] = p;
    lsum += p;
  }
  red[tid] = lsum; __syncthreads();
  for (int st = 64; st > 0; st >>= 1) {
    if (tid < st) red[tid] += red[tid + st];
    __syncthreads();
  }
  float inv = 1.0f / red[0];
  __syncthreads();

  float acc = 0.f;  // weighted time-sum for hidden index tid
  for (int tt = 0; tt < T_DIM; ++tt)
    acc += sbuf[tt] * bf2f(base[(size_t)tt * H_DIM + tid]);
  red[tid] = acc * inv * wf[tid];
  __syncthreads();
  for (int st = 64; st > 0; st >>= 1) {
    if (tid < st) red[tid] += red[tid + st];
    __syncthreads();
  }
  if (tid == 0) outp[b] = 1.0f / (1.0f + __expf(-(red[0] + bfp[0])));
}

extern "C" void kernel_launch(void* const* d_in, const int* in_sizes, int n_in,
                              void* d_out, int out_size, void* d_ws, size_t ws_size,
                              hipStream_t stream) {
  const float* x   = (const float*)d_in[0];  // [B,T,F]
  const float* Wih = (const float*)d_in[1];  // [L,512,128]
  const float* Whh = (const float*)d_in[2];  // [L,512,128]
  const float* bih = (const float*)d_in[3];  // [L,512]
  const float* bhh = (const float*)d_in[4];  // [L,512]
  const float* Wa  = (const float*)d_in[5];  // [A,H]
  const float* ba  = (const float*)d_in[6];  // [A]
  const float* Wf  = (const float*)d_in[7];  // [1,H]
  const float* bf  = (const float*)d_in[8];  // [1]
  float* out = (float*)d_out;                // [B,1]

  const size_t seq_elems = (size_t)B_DIM * T_DIM * H_DIM;  // 16.7M
  unsigned short* x_bf = (unsigned short*)d_ws;            // 33.6 MB
  unsigned short* ping = x_bf + seq_elems;                 // 33.6 MB
  unsigned short* pong = ping + seq_elems;                 // 33.6 MB

  const size_t wstride = (size_t)4 * H_DIM * H_DIM;        // 512*128
  const size_t bstride = (size_t)4 * H_DIM;                // 512

  const int n = (int)seq_elems;
  x_to_bf16_kernel<<<n / 1024, 256, 0, stream>>>(x, x_bf, n);

  const unsigned short* cur = x_bf;
  unsigned short* nxt = ping;
  unsigned short* spare = pong;
  for (int l = 0; l < L_DIM; ++l) {
    lstm_scan<<<B_DIM / 16, 256, 0, stream>>>(
        cur, nxt, Wih + (size_t)l * wstride, Whh + (size_t)l * wstride,
        bih + (size_t)l * bstride, bhh + (size_t)l * bstride);
    const unsigned short* done = nxt;
    // after layer 0 stop recycling x_bf; ping-pong between ping/pong
    nxt = (l == 0) ? spare : (unsigned short*)cur;
    cur = done;
  }

  attn_final_kernel<<<B_DIM, 128, 0, stream>>>(cur, Wa, ba, Wf, bf, out);
}
